// MultiheadAttention_8976481649040
// MI455X (gfx1250) — compile-verified, matrix operand order checked by reading, not run on hardware
//
#include <hip/hip_runtime.h>
#include <hip/hip_bf16.h>

// ---------------------------------------------------------------------------
// MHA for B=4, N=1024, D=1024, H=16, DK=64 on gfx1250 (wave32, WMMA bf16)
// ---------------------------------------------------------------------------

typedef __bf16 v16bf __attribute__((ext_vector_type(16)));
typedef float  v8f   __attribute__((ext_vector_type(8)));

union ABPack { uint4 q[2]; v16bf v; };

// packed f32x2 -> bf16x2 (native packed cvt on gfx1250)
static __device__ __forceinline__ unsigned pk2(float lo, float hi) {
#if __has_builtin(__builtin_amdgcn_cvt_pk_bf16_f32)
    auto t = __builtin_amdgcn_cvt_pk_bf16_f32(lo, hi);
    unsigned u = 0;
    __builtin_memcpy(&u, &t, sizeof(t) > 4 ? 4 : sizeof(t));
    return u;
#else
    union { __bf16 b[2]; unsigned u; } x;
    x.b[0] = (__bf16)lo; x.b[1] = (__bf16)hi;
    return x.u;
#endif
}

static __device__ __forceinline__ unsigned short f2bf(float f) {
    union { __bf16 b; unsigned short s; } x;
    x.b = (__bf16)f;
    return x.s;
}

static __device__ __forceinline__ uint4 cvt8(float4 a, float4 b) {
    uint4 r;
    r.x = pk2(a.x, a.y);
    r.y = pk2(a.z, a.w);
    r.z = pk2(b.x, b.y);
    r.w = pk2(b.z, b.w);
    return r;
}

static __device__ __forceinline__ v8f wmma_bf16(v16bf a, v16bf b, v8f c) {
    return __builtin_amdgcn_wmma_f32_16x16x32_bf16(false, a, false, b,
                                                   (short)0, c, false, false);
}

static __device__ __forceinline__ void async_b128(unsigned lds_addr,
                                                  unsigned long long gaddr) {
    asm volatile("global_load_async_to_lds_b128 %0, %1, off"
                 :: "v"(lds_addr), "v"(gaddr) : "memory");
}
template <int NW>
static __device__ __forceinline__ void wait_async_le() {
    asm volatile("s_wait_asynccnt %0" :: "n"(NW) : "memory");
}
static __device__ __forceinline__ void wait_ds0() {
    asm volatile("s_wait_dscnt 0" ::: "memory");
}

// ---------------------------------------------------------------------------
// Kernel 1: Y = X @ W^T + bias, fp32 in -> bf16 out.
// mode 0/1: dst layout [B,H,N,DK]   (Q, K)
// mode 2  : dst layout [B,H,DK,N]   (V transposed for P*V GEMM)
// grid (16, 32), block 128. Wave computes 32x64 tile; block 128x64.
// W staging is software-pipelined: next k-slab is loaded to registers while
// the WMMA burst runs on the current LDS tile.
// ---------------------------------------------------------------------------
__global__ __launch_bounds__(128) void mha_proj_qkv_kernel(
    const float* __restrict__ src, const float* __restrict__ W,
    const float* __restrict__ bias, unsigned short* __restrict__ dst, int mode)
{
    __shared__ __align__(16) unsigned short Wblk[64 * 32];

    const int lane = threadIdx.x & 31;
    const int wv   = threadIdx.x >> 5;
    const int lm   = lane & 15;
    const int hiq  = lane >> 4;
    const int k0   = hiq * 8;
    const int n0   = blockIdx.x * 64;
    const int m0   = blockIdx.y * 128 + wv * 32;

    const int nl = threadIdx.x >> 1;        // staging row (output col)
    const int hf = threadIdx.x & 1;         // staging k-half
    const float* wbase = W + (size_t)(n0 + nl) * 1024 + hf * 16;
    uint4* wdst = (uint4*)&Wblk[nl * 32 + hf * 16];

    v8f C[2][4];
    #pragma unroll
    for (int rb = 0; rb < 2; ++rb)
        #pragma unroll
        for (int j = 0; j < 4; ++j)
            #pragma unroll
            for (int r = 0; r < 8; ++r) C[rb][j][r] = 0.0f;

    // prologue: stage first W tile
    {
        const float4* ws = (const float4*)wbase;
        float4 a = ws[0], b = ws[1], c = ws[2], d = ws[3];
        wdst[0] = cvt8(a, b);
        wdst[1] = cvt8(c, d);
    }

    for (int kk = 0; kk < 1024; kk += 32) {
        __syncthreads();                    // current LDS tile visible

        const bool more = (kk + 32) < 1024;
        float4 na, nb, nc, nd;
        if (more) {                         // overlap next W slab with compute
            const float4* ws = (const float4*)(wbase + kk + 32);
            na = ws[0]; nb = ws[1]; nc = ws[2]; nd = ws[3];
        }

        v16bf A[2];
        #pragma unroll
        for (int rb = 0; rb < 2; ++rb) {
            const float* arow = src + (size_t)(m0 + rb * 16 + lm) * 1024 + kk;
            const float4* p0 = (const float4*)(arow + k0);
            const float4* p1 = (const float4*)(arow + k0 + 16);
            ABPack p;
            p.q[0] = cvt8(p0[0], p0[1]);
            p.q[1] = cvt8(p1[0], p1[1]);
            A[rb] = p.v;
        }
        ABPack bp[4];
        #pragma unroll
        for (int j = 0; j < 4; ++j) {
            const unsigned short* br = &Wblk[(j * 16 + lm) * 32 + k0];
            bp[j].q[0] = *(const uint4*)br;
            bp[j].q[1] = *(const uint4*)(br + 16);
        }
        #pragma unroll
        for (int j = 0; j < 4; ++j)
            #pragma unroll
            for (int rb = 0; rb < 2; ++rb)
                C[rb][j] = wmma_bf16(A[rb], bp[j].v, C[rb][j]);

        __syncthreads();                    // all reads of LDS tile done
        if (more) {
            wdst[0] = cvt8(na, nb);
            wdst[1] = cvt8(nc, nd);
        }
    }

    #pragma unroll
    for (int j = 0; j < 4; ++j) {
        const int n = n0 + j * 16 + lm;
        const float bn = bias[n];
        const int h  = n >> 6;
        const int dk = n & 63;
        #pragma unroll
        for (int rb = 0; rb < 2; ++rb) {
            #pragma unroll
            for (int r = 0; r < 8; ++r) {
                int m = m0 + rb * 16 + r + 8 * hiq;
                int b  = m >> 10;
                int nr = m & 1023;
                float v = C[rb][j][r] + bn;
                size_t idx = (mode == 2)
                    ? ((size_t)((b * 16 + h) * 64 + dk)) * 1024 + nr
                    : ((size_t)((b * 16 + h) * 1024 + nr)) * 64 + dk;
                dst[idx] = f2bf(v);
            }
        }
    }
}

// ---------------------------------------------------------------------------
// Kernel 2: flash attention per (b,h). grid (N/64, B*H), block 128 (4 waves).
// K/V tiles double-buffered in LDS with gfx1250 async global->LDS loads:
// next tile's loads are issued before computing the current tile, then
// s_wait_asynccnt 4 retires only the older group (async loads complete
// in order per thread).
// ---------------------------------------------------------------------------
__global__ __launch_bounds__(128) void mha_attn_kernel(
    const unsigned short* __restrict__ Qh, const unsigned short* __restrict__ Kh,
    const unsigned short* __restrict__ Vt, const float* __restrict__ sph,
    const unsigned char* __restrict__ mask, unsigned short* __restrict__ Xa)
{
    __shared__ __align__(16) unsigned short Kblk[2][32 * 64];   // [buf][key][d]
    __shared__ __align__(16) unsigned short Vtblk[2][64 * 32];  // [buf][d][key]
    __shared__ __align__(16) unsigned short Pbuf[4][16 * 32];

    const int lane = threadIdx.x & 31;
    const int wv   = threadIdx.x >> 5;
    const int lm   = lane & 15;
    const int hiq  = lane >> 4;
    const int k0   = hiq * 8;
    const int bh   = blockIdx.y;
    const int b    = bh >> 4;
    const int h    = bh & 15;
    const int q0   = blockIdx.x * 64;

    const unsigned short* Qb = Qh + (size_t)bh * 1024 * 64;
    const unsigned short* Kb = Kh + (size_t)bh * 1024 * 64;
    const unsigned short* Vb = Vt + (size_t)bh * 64 * 1024;
    const float*          sb = sph + (size_t)b * 1024 * 1024;
    const unsigned char*  mb = mask + b * 1024;

    auto stage = [&](int bufi, int kbv) {
        int t = threadIdx.x;
        unsigned ldsk = (unsigned)(unsigned long long)(void*)&Kblk[bufi][t * 16];
        unsigned long long gk =
            (unsigned long long)(Kb + (size_t)kbv * 64 + t * 16);
        async_b128(ldsk, gk);
        async_b128(ldsk + 16, gk + 16);

        int d = t >> 1, hf = t & 1;
        unsigned ldsv =
            (unsigned)(unsigned long long)(void*)&Vtblk[bufi][d * 32 + hf * 16];
        unsigned long long gv =
            (unsigned long long)(Vb + (size_t)d * 1024 + kbv + hf * 16);
        async_b128(ldsv, gv);
        async_b128(ldsv + 16, gv + 16);
    };

    // Preload this wave's 16x64 Q tile in A-matrix layout (two K=32 slabs).
    v16bf AQ[2];
    {
        const unsigned short* qrow = Qb + (size_t)(q0 + wv * 16 + lm) * 64;
        #pragma unroll
        for (int dd = 0; dd < 2; ++dd) {
            ABPack p;
            p.q[0] = *(const uint4*)(qrow + dd * 32 + k0);
            p.q[1] = *(const uint4*)(qrow + dd * 32 + k0 + 16);
            AQ[dd] = p.v;
        }
    }

    float mrow[8], lrow[8];
    v8f O[4];
    #pragma unroll
    for (int r = 0; r < 8; ++r) { mrow[r] = -INFINITY; lrow[r] = 0.0f; }
    #pragma unroll
    for (int j = 0; j < 4; ++j)
        #pragma unroll
        for (int r = 0; r < 8; ++r) O[j][r] = 0.0f;

    stage(0, 0);                            // prologue fill of buffer 0

    for (int it = 0; it < 32; ++it) {
        const int kb  = it * 32;
        const int cur = it & 1;

        __syncthreads();                    // prior reads of next buffer done
        const bool more = (it + 1) < 32;
        if (more) {
            stage(cur ^ 1, kb + 32);        // overlap next fill with compute
            wait_async_le<4>();             // older group (current tile) done
        } else {
            wait_async_le<0>();
        }
        __syncthreads();                    // whole workgroup's tile ready

        // S = Q * K^T for two 16-key tiles, then scale * sph, key mask
        float S[2][8];
        #pragma unroll
        for (int t = 0; t < 2; ++t) {
            v8f Ct;
            #pragma unroll
            for (int r = 0; r < 8; ++r) Ct[r] = 0.0f;
            const int keyl = t * 16 + lm;
            ABPack kp[2];
            #pragma unroll
            for (int dd = 0; dd < 2; ++dd) {
                const unsigned short* kr = &Kblk[cur][keyl * 64 + dd * 32 + k0];
                kp[dd].q[0] = *(const uint4*)kr;
                kp[dd].q[1] = *(const uint4*)(kr + 16);
            }
            #pragma unroll
            for (int dd = 0; dd < 2; ++dd)
                Ct = wmma_bf16(AQ[dd], kp[dd].v, Ct);

            const int kg = kb + keyl;
            const bool ism = mb[kg] != 0;
            #pragma unroll
            for (int r = 0; r < 8; ++r) {
                int q = q0 + wv * 16 + r + 8 * hiq;
                float s = Ct[r] * 0.125f * sb[(size_t)q * 1024 + kg];
                S[t][r] = ism ? -1.0e30f : s;
            }
        }

        // online softmax update (rows live across 16-lane groups)
        float alpha[8];
        #pragma unroll
        for (int r = 0; r < 8; ++r) {
            float x = fmaxf(S[0][r], S[1][r]);
            #pragma unroll
            for (int sm = 8; sm >= 1; sm >>= 1)
                x = fmaxf(x, __shfl_xor(x, sm, 16));
            float mn = fmaxf(mrow[r], x);
            float p0 = __expf(S[0][r] - mn);
            float p1 = __expf(S[1][r] - mn);
            float rs = p0 + p1;
            #pragma unroll
            for (int sm = 8; sm >= 1; sm >>= 1)
                rs += __shfl_xor(rs, sm, 16);
            alpha[r] = __expf(mrow[r] - mn);
            lrow[r]  = lrow[r] * alpha[r] + rs;
            mrow[r]  = mn;
            Pbuf[wv][(r + 8 * hiq) * 32 + lm]      = f2bf(p0);
            Pbuf[wv][(r + 8 * hiq) * 32 + 16 + lm] = f2bf(p1);
        }
        #pragma unroll
        for (int j = 0; j < 4; ++j)
            #pragma unroll
            for (int r = 0; r < 8; ++r) O[j][r] *= alpha[r];

        wait_ds0();   // P writes visible before the A-layout reads below

        // O += P * V  (A = P 16x32, B = Vt slice 32x16 per 16-d column block)
        ABPack pa;
        {
            const unsigned short* pr = &Pbuf[wv][lm * 32 + k0];
            pa.q[0] = *(const uint4*)pr;
            pa.q[1] = *(const uint4*)(pr + 16);
        }
        ABPack vp[4];
        #pragma unroll
        for (int j = 0; j < 4; ++j) {
            const unsigned short* vr = &Vtblk[cur][(j * 16 + lm) * 32 + k0];
            vp[j].q[0] = *(const uint4*)vr;
            vp[j].q[1] = *(const uint4*)(vr + 16);
        }
        #pragma unroll
        for (int j = 0; j < 4; ++j)
            O[j] = wmma_bf16(pa.v, vp[j].v, O[j]);
    }

    // normalize and write bf16 attention output in [B,N,D] layout
    #pragma unroll
    for (int j = 0; j < 4; ++j) {
        #pragma unroll
        for (int r = 0; r < 8; ++r) {
            int q = q0 + wv * 16 + r + 8 * hiq;
            float v = O[j][r] / lrow[r];
            Xa[(size_t)(b * 1024 + q) * 1024 + h * 64 + j * 16 + lm] = f2bf(v);
        }
    }
}

// ---------------------------------------------------------------------------
// Kernel 3: out = Xa(bf16) @ Wo^T + bo, fp32 out. Same pipelined tiling.
// ---------------------------------------------------------------------------
__global__ __launch_bounds__(128) void mha_out_proj_kernel(
    const unsigned short* __restrict__ X, const float* __restrict__ W,
    const float* __restrict__ bias, float* __restrict__ out)
{
    __shared__ __align__(16) unsigned short Wblk[64 * 32];

    const int lane = threadIdx.x & 31;
    const int wv   = threadIdx.x >> 5;
    const int lm   = lane & 15;
    const int hiq  = lane >> 4;
    const int k0   = hiq * 8;
    const int n0   = blockIdx.x * 64;
    const int m0   = blockIdx.y * 128 + wv * 32;

    const int nl = threadIdx.x >> 1;
    const int hf = threadIdx.x & 1;
    const float* wbase = W + (size_t)(n0 + nl) * 1024 + hf * 16;
    uint4* wdst = (uint4*)&Wblk[nl * 32 + hf * 16];

    v8f C[2][4];
    #pragma unroll
    for (int rb = 0; rb < 2; ++rb)
        #pragma unroll
        for (int j = 0; j < 4; ++j)
            #pragma unroll
            for (int r = 0; r < 8; ++r) C[rb][j][r] = 0.0f;

    {
        const float4* ws = (const float4*)wbase;
        float4 a = ws[0], b = ws[1], c = ws[2], d = ws[3];
        wdst[0] = cvt8(a, b);
        wdst[1] = cvt8(c, d);
    }

    for (int kk = 0; kk < 1024; kk += 32) {
        __syncthreads();

        const bool more = (kk + 32) < 1024;
        float4 na, nb, nc, nd;
        if (more) {
            const float4* ws = (const float4*)(wbase + kk + 32);
            na = ws[0]; nb = ws[1]; nc = ws[2]; nd = ws[3];
        }

        v16bf A[2];
        #pragma unroll
        for (int rb = 0; rb < 2; ++rb) {
            const unsigned short* arow =
                X + (size_t)(m0 + rb * 16 + lm) * 1024 + kk;
            ABPack p;
            p.q[0] = *(const uint4*)(arow + k0);
            p.q[1] = *(const uint4*)(arow + k0 + 16);
            A[rb] = p.v;
        }
        ABPack bp[4];
        #pragma unroll
        for (int j = 0; j < 4; ++j) {
            const unsigned short* br = &Wblk[(j * 16 + lm) * 32 + k0];
            bp[j].q[0] = *(const uint4*)br;
            bp[j].q[1] = *(const uint4*)(br + 16);
        }
        #pragma unroll
        for (int j = 0; j < 4; ++j)
            #pragma unroll
            for (int rb = 0; rb < 2; ++rb)
                C[rb][j] = wmma_bf16(A[rb], bp[j].v, C[rb][j]);

        __syncthreads();
        if (more) {
            wdst[0] = cvt8(na, nb);
            wdst[1] = cvt8(nc, nd);
        }
    }

    #pragma unroll
    for (int j = 0; j < 4; ++j) {
        const int n = n0 + j * 16 + lm;
        const float bn = bias[n];
        #pragma unroll
        for (int rb = 0; rb < 2; ++rb) {
            #pragma unroll
            for (int r = 0; r < 8; ++r) {
                int m = m0 + rb * 16 + r + 8 * hiq;
                out[(size_t)m * 1024 + n] = C[rb][j][r] + bn;
            }
        }
    }
}

// ---------------------------------------------------------------------------
extern "C" void kernel_launch(void* const* d_in, const int* in_sizes, int n_in,
                              void* d_out, int out_size, void* d_ws, size_t ws_size,
                              hipStream_t stream) {
    (void)in_sizes; (void)n_in; (void)out_size; (void)ws_size;

    const float* query = (const float*)d_in[0];
    const float* key_  = (const float*)d_in[1];
    const float* value = (const float*)d_in[2];
    const float* sph   = (const float*)d_in[3];
    const unsigned char* mask = (const unsigned char*)d_in[4];  // jax bool = 1B
    const float* Wq = (const float*)d_in[5];
    const float* bq = (const float*)d_in[6];
    const float* Wk = (const float*)d_in[7];
    const float* bk = (const float*)d_in[8];
    const float* Wv = (const float*)d_in[9];
    const float* bv = (const float*)d_in[10];
    const float* Wo = (const float*)d_in[11];
    const float* bo = (const float*)d_in[12];

    // workspace: Qh, Kh (B,H,N,DK), Vt (B,H,DK,N), Xa (B,N,D) — all bf16
    const size_t elems = (size_t)4 * 16 * 1024 * 64;   // 4,194,304
    unsigned short* Qh = (unsigned short*)d_ws;
    unsigned short* Kh = Qh + elems;
    unsigned short* Vt = Kh + elems;
    unsigned short* Xa = Vt + elems;

    dim3 blk(128);
    mha_proj_qkv_kernel<<<dim3(16, 32), blk, 0, stream>>>(query, Wq, bq, Qh, 0);
    mha_proj_qkv_kernel<<<dim3(16, 32), blk, 0, stream>>>(key_,  Wk, bk, Kh, 1);
    mha_proj_qkv_kernel<<<dim3(16, 32), blk, 0, stream>>>(value, Wv, bv, Vt, 2);
    mha_attn_kernel<<<dim3(16, 64), blk, 0, stream>>>(Qh, Kh, Vt, sph, mask, Xa);
    mha_out_proj_kernel<<<dim3(16, 32), blk, 0, stream>>>(Xa, Wo, bo, (float*)d_out);
}
